// Net_12996571038297
// MI455X (gfx1250) — matrix operand order repaired and checked by reading, not compiled, not run on hardware
//
#include <hip/hip_runtime.h>
#include <hip/hip_bf16.h>

// ---------------------------------------------------------------------------
// CDNA5 (gfx1250) fused EdgeConv network.
// Wave32, v_wmma_f32_16x16x32_f16 for all edge-MLP GEMMs.
// Each block: 64 edges; gather -> 3 WMMA GEMM stages in LDS -> atomic scatter.
// ---------------------------------------------------------------------------

typedef _Float16 v16h __attribute__((ext_vector_type(16)));
typedef _Float16 v8h  __attribute__((ext_vector_type(8)));
typedef float    v8f  __attribute__((ext_vector_type(8)));

union HU { v16h v; v8h h[2]; };

// Fused EdgeConv: for a tile of 64 edges, build [x_i, x_j - x_i] in LDS,
// run the 3-layer MLP with WMMA entirely in LDS, atomically scatter-add the
// result into per-node fp32 accumulators (segment_sum over dst).
//   FIN : input node feature dim (3 means "read fp32 x directly")
//   F   : hidden/output dim of this conv's MLP
//   KIN : padded K of the first GEMM (2*FIN padded up to multiple of 32)
// Wave work split: 8 waves; wave owns row tile (wave&3), column tiles
// ct = (wave>>2) + 2*j for j in [0, F/32).  A fragments are register-cached
// per wave and reused across all its column tiles.
template <int FIN, int F, int KIN>
__global__ __launch_bounds__(256) void edgeconv_kernel(
    const float* __restrict__ x,          // [N,3] fp32 (layer 1 only)
    const _Float16* __restrict__ hin,     // [N,FIN] f16 (layers 2,3)
    const int* __restrict__ src,
    const int* __restrict__ dst,
    const _Float16* __restrict__ W1, const float* __restrict__ b1,  // Wt[F][KIN]
    const _Float16* __restrict__ W2, const float* __restrict__ b2,  // Wt[F][F]
    const _Float16* __restrict__ W3, const float* __restrict__ b3,  // Wt[F][F]
    float* __restrict__ acc,              // [N,F] fp32 accumulators
    int E) {
  constexpr int LDA  = (KIN > F ? KIN : F);
  constexpr int LDAP = LDA + 8;   // +16B skew: row r starts 4 banks after r-1
  constexpr int LDBP = F + 8;
  constexpr int CT   = F / 16;    // column tiles
  constexpr int CPW  = CT / 2;    // column tiles per wave
  constexpr int KT1  = KIN / 32;  // K-steps, stage 1
  constexpr int KT23 = F / 32;    // K-steps, stages 2/3
  extern __shared__ char smem[];
  _Float16* bufA = (_Float16*)smem;            // [64][LDAP]
  _Float16* bufB = bufA + 64 * LDAP;           // [64][LDBP]
  int* sdst = (int*)(bufB + 64 * LDBP);        // [64]

  const int tid  = threadIdx.x;
  const int lane = tid & 31;
  const int wave = tid >> 5;
  const int r16  = lane & 15;
  const int hi   = lane >> 4;
  const int h8   = hi * 8;
  const int hi16 = hi * 16;
  const int rowBase = (wave & 3) * 16;  // this wave's fixed 16-edge row tile
  const int cw      = wave >> 2;        // column-tile phase (0 or 1)
  const int eBase   = blockIdx.x * 64;

  // Warm L2/WGP$ for the weight streams (global_prefetch_b8).
  {
    const int pfo = tid * 256;
    if (pfo < F * KIN * 2) __builtin_prefetch((const char*)W1 + pfo, 0, 1);
    if (pfo < F * F * 2) {
      __builtin_prefetch((const char*)W2 + pfo, 0, 1);
      __builtin_prefetch((const char*)W3 + pfo, 0, 1);
    }
  }

  if (tid < 64) sdst[tid] = dst[eBase + tid];

  // Gather: bufA[row][c] = (c < FIN) ? h[dst][c] : h[src][c-FIN]-h[dst][c-FIN]
  for (int idx = tid; idx < 64 * KIN; idx += 256) {
    const int row = idx / KIN;
    const int c   = idx & (KIN - 1);
    const int e   = eBase + row;
    const int s   = src[e];
    const int d   = dst[e];
    _Float16 val;
    if (FIN == 3) {
      if (c < 3)      val = (_Float16)x[d * 3 + c];
      else if (c < 6) val = (_Float16)(x[s * 3 + (c - 3)] - x[d * 3 + (c - 3)]);
      else            val = (_Float16)0.f;
    } else {
      if (c < FIN) {
        val = hin[d * FIN + c];
      } else {
        const int cc = c - FIN;
        val = (_Float16)((float)hin[s * FIN + cc] - (float)hin[d * FIN + cc]);
      }
    }
    bufA[row * LDAP + c] = val;
  }
  __syncthreads();

  // ---- stage 1: C1 = relu(A0 * W1 + b1), K = KIN, bufA -> bufB ----
  {
    const _Float16* arow = bufA + (rowBase + r16) * LDAP;
    HU af[KT1];
#pragma unroll
    for (int kt = 0; kt < KT1; ++kt) {
      af[kt].h[0] = *(const v8h*)(arow + kt * 32 + h8);
      af[kt].h[1] = *(const v8h*)(arow + kt * 32 + 16 + h8);
    }
    for (int j = 0; j < CPW; ++j) {
      const int n = (cw + 2 * j) * 16 + r16;
      const _Float16* brow = W1 + n * KIN;
      v8f c = {};
#pragma unroll
      for (int kt = 0; kt < KT1; ++kt) {
        HU bu; bu.h[0] = *(const v8h*)(brow + kt * 32 + h8); bu.h[1] = *(const v8h*)(brow + kt * 32 + 16 + h8); v16h b = bu.v;
        c = __builtin_amdgcn_wmma_f32_16x16x32_f16(false, af[kt].v, false, b,
                                                   (short)0, c, false, false);
      }
      const float bv = b1[n];
#pragma unroll
      for (int v = 0; v < 8; ++v) {
        float val = c[v] + bv;
        val = val > 0.f ? val : 0.f;
        bufB[(rowBase + v + 8 * hi) * LDBP + n] = (_Float16)val;
      }
    }
  }
  __syncthreads();

  // ---- stage 2: C2 = relu(C1 * W2 + b2), K = F, bufB -> bufA ----
  {
    const _Float16* arow = bufB + (rowBase + r16) * LDBP;
    HU af[KT23];
#pragma unroll
    for (int kt = 0; kt < KT23; ++kt) {
      af[kt].h[0] = *(const v8h*)(arow + kt * 32 + h8);
      af[kt].h[1] = *(const v8h*)(arow + kt * 32 + 16 + h8);
    }
    for (int j = 0; j < CPW; ++j) {
      const int n = (cw + 2 * j) * 16 + r16;
      const _Float16* brow = W2 + n * F;
      v8f c = {};
#pragma unroll
      for (int kt = 0; kt < KT23; ++kt) {
        HU bu; bu.h[0] = *(const v8h*)(brow + kt * 32 + h8); bu.h[1] = *(const v8h*)(brow + kt * 32 + 16 + h8); v16h b = bu.v;
        c = __builtin_amdgcn_wmma_f32_16x16x32_f16(false, af[kt].v, false, b,
                                                   (short)0, c, false, false);
      }
      const float bv = b2[n];
#pragma unroll
      for (int v = 0; v < 8; ++v) {
        float val = c[v] + bv;
        val = val > 0.f ? val : 0.f;
        bufA[(rowBase + v + 8 * hi) * LDAP + n] = (_Float16)val;
      }
    }
  }
  __syncthreads();

  // ---- stage 3: M = C2 * W3 + b3, scatter-add into acc[dst] ----
  {
    const _Float16* arow = bufA + (rowBase + r16) * LDAP;
    HU af[KT23];
#pragma unroll
    for (int kt = 0; kt < KT23; ++kt) {
      af[kt].h[0] = *(const v8h*)(arow + kt * 32 + h8);
      af[kt].h[1] = *(const v8h*)(arow + kt * 32 + 16 + h8);
    }
    for (int j = 0; j < CPW; ++j) {
      const int n = (cw + 2 * j) * 16 + r16;
      const _Float16* brow = W3 + n * F;
      v8f c = {};
#pragma unroll
      for (int kt = 0; kt < KT23; ++kt) {
        HU bu; bu.h[0] = *(const v8h*)(brow + kt * 32 + h8); bu.h[1] = *(const v8h*)(brow + kt * 32 + 16 + h8); v16h b = bu.v;
        c = __builtin_amdgcn_wmma_f32_16x16x32_f16(false, af[kt].v, false, b,
                                                   (short)0, c, false, false);
      }
      const float bv = b3[n];
#pragma unroll
      for (int v = 0; v < 8; ++v) {
        const int node = sdst[rowBase + v + 8 * hi];
        atomicAdd(acc + node * F + n, c[v] + bv);
      }
    }
  }
}

// Transpose + f16-convert + K-pad a weight matrix W[K][N] -> Wt[N][Kpad].
__global__ void wprep_kernel(const float* __restrict__ W,
                             _Float16* __restrict__ Wt,
                             int K, int N, int Kpad) {
  const int idx = blockIdx.x * blockDim.x + threadIdx.x;
  if (idx >= N * Kpad) return;
  const int n = idx / Kpad;
  const int k = idx - n * Kpad;
  Wt[idx] = (k < K) ? (_Float16)W[k * N + n] : (_Float16)0.f;
}

__global__ void zero_f32_kernel(float* __restrict__ p, int n) {
  const int i = blockIdx.x * blockDim.x + threadIdx.x;
  if (i < n) p[i] = 0.f;
}

__global__ void f32_to_f16_kernel(const float* __restrict__ in,
                                  _Float16* __restrict__ out, int n) {
  const int i = blockIdx.x * blockDim.x + threadIdx.x;
  if (i < n) out[i] = (_Float16)in[i];
}

// Segment-sum node features into per-graph sums + node counts.
__global__ void pool_kernel(const float* __restrict__ acc,
                            const int* __restrict__ batch,
                            float* __restrict__ sums,
                            float* __restrict__ cnt, int Nn, int F) {
  const int i = blockIdx.x * blockDim.x + threadIdx.x;
  if (i >= Nn * F) return;
  const int node = i / F;
  const int c = i - node * F;
  const int b = batch[node];
  atomicAdd(sums + b * F + c, acc[i]);
  if (c == 0) atomicAdd(cnt + b, 1.f);
}

// Per-graph head: g = sums/max(cnt,1); h = g@l2_w+l2_b; out = sigmoid(h@l3_w+l3_b)
__global__ __launch_bounds__(64) void head_kernel(
    const float* __restrict__ sums, const float* __restrict__ cnt,
    const float* __restrict__ l2w, const float* __restrict__ l2b,
    const float* __restrict__ l3w, const float* __restrict__ l3b,
    float* __restrict__ out) {
  __shared__ float g[256];
  __shared__ float h1[64];
  const int gi = blockIdx.x;
  float c = cnt[gi];
  c = c > 1.f ? c : 1.f;
  for (int k = threadIdx.x; k < 256; k += 64) g[k] = sums[gi * 256 + k] / c;
  __syncthreads();
  const int j = threadIdx.x;
  float a = l2b[j];
  for (int k = 0; k < 256; ++k) a += g[k] * l2w[k * 64 + j];
  h1[j] = a;
  __syncthreads();
  if (j < 2) {
    float o = l3b[j];
    for (int k = 0; k < 64; ++k) o += h1[k] * l3w[k * 2 + j];
    out[gi * 2 + j] = 1.f / (1.f + expf(-o));
  }
}

extern "C" void kernel_launch(void* const* d_in, const int* in_sizes, int n_in,
                              void* d_out, int out_size, void* d_ws,
                              size_t ws_size, hipStream_t stream) {
  (void)in_sizes; (void)n_in; (void)out_size; (void)ws_size;
  const int N = 10000, E = 320000, G = 64;

  const float* x     = (const float*)d_in[0];
  const int*   ei    = (const int*)d_in[1];
  const int*   batch = (const int*)d_in[2];
  const float* c1w1 = (const float*)d_in[3];  const float* c1b1 = (const float*)d_in[4];
  const float* c1w2 = (const float*)d_in[5];  const float* c1b2 = (const float*)d_in[6];
  const float* c1w3 = (const float*)d_in[7];  const float* c1b3 = (const float*)d_in[8];
  const float* c2w1 = (const float*)d_in[9];  const float* c2b1 = (const float*)d_in[10];
  const float* c2w2 = (const float*)d_in[11]; const float* c2b2 = (const float*)d_in[12];
  const float* c2w3 = (const float*)d_in[13]; const float* c2b3 = (const float*)d_in[14];
  const float* c3w1 = (const float*)d_in[15]; const float* c3b1 = (const float*)d_in[16];
  const float* c3w2 = (const float*)d_in[17]; const float* c3b2 = (const float*)d_in[18];
  const float* c3w3 = (const float*)d_in[19]; const float* c3b3 = (const float*)d_in[20];
  const float* l2w  = (const float*)d_in[21]; const float* l2b  = (const float*)d_in[22];
  const float* l3w  = (const float*)d_in[23]; const float* l3b  = (const float*)d_in[24];

  const int* src = ei;
  const int* dst = ei + E;

  // ---- workspace carve-out (~16 MB) ----
  char* ws = (char*)d_ws;
  size_t off = 0;
  auto carve = [&](size_t bytes) {
    char* p = ws + off;
    off += (bytes + 255) & ~(size_t)255;
    return p;
  };
  _Float16* wt11 = (_Float16*)carve(64 * 32 * 2);
  _Float16* wt12 = (_Float16*)carve(64 * 64 * 2);
  _Float16* wt13 = (_Float16*)carve(64 * 64 * 2);
  _Float16* wt21 = (_Float16*)carve(128 * 128 * 2);
  _Float16* wt22 = (_Float16*)carve(128 * 128 * 2);
  _Float16* wt23 = (_Float16*)carve(128 * 128 * 2);
  _Float16* wt31 = (_Float16*)carve(256 * 256 * 2);
  _Float16* wt32 = (_Float16*)carve(256 * 256 * 2);
  _Float16* wt33 = (_Float16*)carve(256 * 256 * 2);
  _Float16* h16  = (_Float16*)carve((size_t)N * 256 * 2);  // node feats f16
  float*    acc  = (float*)carve((size_t)N * 256 * 4);     // node accum f32
  float*    sums = (float*)carve((size_t)G * 256 * 4);
  float*    cnt  = (float*)carve((size_t)G * 4);

  auto wp = [&](const float* W, _Float16* Wt, int K, int Nc, int Kp) {
    const int tot = Nc * Kp;
    wprep_kernel<<<(tot + 255) / 256, 256, 0, stream>>>(W, Wt, K, Nc, Kp);
  };
  wp(c1w1, wt11, 6, 64, 32);
  wp(c1w2, wt12, 64, 64, 64);
  wp(c1w3, wt13, 64, 64, 64);
  wp(c2w1, wt21, 128, 128, 128);
  wp(c2w2, wt22, 128, 128, 128);
  wp(c2w3, wt23, 128, 128, 128);
  wp(c3w1, wt31, 256, 256, 256);
  wp(c3w2, wt32, 256, 256, 256);
  wp(c3w3, wt33, 256, 256, 256);

  auto zero = [&](float* p, int n) {
    zero_f32_kernel<<<(n + 255) / 256, 256, 0, stream>>>(p, n);
  };
  auto shbytes = [](int KIN, int F) {
    const int LDA = KIN > F ? KIN : F;
    return (size_t)(64 * (LDA + 8) + 64 * (F + 8)) * 2 + 64 * 4;
  };
  const int eBlocks = E / 64;  // E is a multiple of 64

  // ---- conv1: MLP(6->64->64->64) ----
  zero(acc, N * 64);
  edgeconv_kernel<3, 64, 32><<<eBlocks, 256, shbytes(32, 64), stream>>>(
      x, h16, src, dst, wt11, c1b1, wt12, c1b2, wt13, c1b3, acc, E);
  f32_to_f16_kernel<<<(N * 64 + 255) / 256, 256, 0, stream>>>(acc, h16, N * 64);

  // ---- conv2: MLP(128->128->128->128) ----
  zero(acc, N * 128);
  edgeconv_kernel<64, 128, 128><<<eBlocks, 256, shbytes(128, 128), stream>>>(
      x, h16, src, dst, wt21, c2b1, wt22, c2b2, wt23, c2b3, acc, E);
  f32_to_f16_kernel<<<(N * 128 + 255) / 256, 256, 0, stream>>>(acc, h16, N * 128);

  // ---- conv3: MLP(256->256->256->256) ----
  zero(acc, N * 256);
  edgeconv_kernel<128, 256, 256><<<eBlocks, 256, shbytes(256, 256), stream>>>(
      x, h16, src, dst, wt31, c3b1, wt32, c3b2, wt33, c3b3, acc, E);

  // ---- global mean pool + head ----
  zero(sums, G * 256);
  zero(cnt, G);
  pool_kernel<<<(N * 256 + 255) / 256, 256, 0, stream>>>(acc, batch, sums, cnt,
                                                         N, 256);
  head_kernel<<<G, 64, 0, stream>>>(sums, cnt, l2w, l2b, l3w, l3b,
                                    (float*)d_out);
}
